// DecoderRNN_23012434772082
// MI455X (gfx1250) — matrix operand order repaired and checked by reading, not compiled, not run on hardware
//
#include <hip/hip_runtime.h>
#include <math.h>

#define H 1024
#define V 2048
#define B 256
#define S 512
#define SOS 129
#define NSTEPS 6

typedef __attribute__((ext_vector_type(16))) __bf16 v16bf;
typedef __attribute__((ext_vector_type(8)))  __bf16 v8bf;
typedef __attribute__((ext_vector_type(8)))  float  v8f;

// Async global->LDS support (gfx1250). Guarded so either toolchain compiles.
#if defined(__gfx1250__) && \
    __has_builtin(__builtin_amdgcn_global_load_async_to_lds_b128) && \
    __has_builtin(__builtin_amdgcn_s_wait_asynccnt)
#define HAS_ASYNC_LDS 1
typedef __attribute__((ext_vector_type(4))) int i32x4;
typedef __attribute__((address_space(1))) i32x4 gbl_i32x4;   // global int4
typedef __attribute__((address_space(3))) i32x4 lds_i32x4;   // LDS int4
#else
#define HAS_ASYNC_LDS 0
#endif

__device__ __forceinline__ float sigmoidf_(float x) { return 1.0f / (1.0f + expf(-x)); }

// A-fragment per ISA 7.12.2: elements 0..7 = K kbase..+7, 8..15 = K kbase+16..+23
__device__ __forceinline__ v16bf load_afrag(const __bf16* p) {
    v8bf lo = *(const v8bf*)(p);
    v8bf hi = *(const v8bf*)(p + 16);
    return __builtin_shufflevector(lo, hi, 0,1,2,3,4,5,6,7,8,9,10,11,12,13,14,15);
}

// ---------------------------------------------------------------- conversions
__global__ void f32_to_bf16_kernel(const float* __restrict__ in, __bf16* __restrict__ out, int n) {
    int i = blockIdx.x * blockDim.x + threadIdx.x;
    if (i < n) out[i] = (__bf16)in[i];
}

// ------------------------------------------------------------ embedding gather
__global__ void embed_kernel(const float* __restrict__ emb, const int* __restrict__ inp,
                             int use_sos, __bf16* __restrict__ x) {
    int i = blockIdx.x * blockDim.x + threadIdx.x;     // B*H threads
    int b = i >> 10;
    int h = i & (H - 1);
    int idx = use_sos ? SOS : inp[b];
    x[i] = (__bf16)emb[(size_t)idx * H + h];
}

// -------------------------------------------------------------- initial state
__global__ void init_state_kernel(const float* __restrict__ h0, const float* __restrict__ c0,
                                  float* __restrict__ hF, float* __restrict__ cF,
                                  __bf16* __restrict__ hB) {
    int i = blockIdx.x * blockDim.x + threadIdx.x;     // B*H
    float h = h0[i];
    hF[i] = h;
    cF[i] = c0[i];
    hB[i] = (__bf16)h;
}

// --------------------------------------------------------------- bf16 WMMA GEMM
// C[M,N] = act( A1[M,K1] @ W1[N,K1]^T + A2[M,K2] @ W2[N,K2]^T + bias1 + bias2 )
// M == B == 256. One wave computes a 64x16 tile: 4 M-subtiles reuse one W
// fragment per K-chunk (weights are the large operand -> 4x fewer W fetches).
__global__ __launch_bounds__(256)
void wmma_gemm_bf16(const __bf16* __restrict__ A1, int lda1,
                    const __bf16* __restrict__ W1, int ldw1, int K1,
                    const __bf16* __restrict__ A2, int lda2,
                    const __bf16* __restrict__ W2, int ldw2, int K2,
                    const float* __restrict__ bias1, const float* __restrict__ bias2,
                    float* __restrict__ outF, int ldf,
                    __bf16* __restrict__ outB, int ldb_,
                    int N, int act) {
    const int MG = B / 64;                              // 4 groups of 64 rows
    int gwave = (blockIdx.x * blockDim.x + threadIdx.x) >> 5;
    int lane  = threadIdx.x & 31;
    int mg = gwave % MG;
    int nt = gwave / MG;
    if (nt >= (N >> 4)) return;                         // wave-uniform

    int rowInTile = lane & 15;
    int khalf     = lane >> 4;
    int ncol      = nt * 16 + rowInTile;

    v8f acc[4] = {{}, {}, {}, {}};

    auto run_pair = [&](const __bf16* A, int lda, const __bf16* W, int ldw, int K) {
        const __bf16* aRow[4];
#pragma unroll
        for (int j = 0; j < 4; ++j)
            aRow[j] = A + (size_t)(mg * 64 + j * 16 + rowInTile) * lda + khalf * 8;
        const __bf16* wRow = W + (size_t)ncol * ldw + khalf * 8;
        for (int k0 = 0; k0 < K; k0 += 32) {
            v16bf bfrag = load_afrag(wRow + k0);
#pragma unroll
            for (int j = 0; j < 4; ++j) {
                v16bf afrag = load_afrag(aRow[j] + k0);
                acc[j] = __builtin_amdgcn_wmma_f32_16x16x32_bf16(
                             false, afrag, false, bfrag, (short)0, acc[j], false, false);
            }
        }
    };

    run_pair(A1, lda1, W1, ldw1, K1);
    if (A2) run_pair(A2, lda2, W2, ldw2, K2);

    float bsum = (bias1 ? bias1[ncol] : 0.0f) + (bias2 ? bias2[ncol] : 0.0f);
#pragma unroll
    for (int j = 0; j < 4; ++j) {
#pragma unroll
        for (int r = 0; r < 8; ++r) {
            int m = mg * 64 + j * 16 + khalf * 8 + r;   // ISA C/D layout
            float v = acc[j][r] + bsum;
            if (act == 1) v = tanhf(v);
            if (outF) outF[(size_t)m * ldf + ncol] = v;
            if (outB) outB[(size_t)m * ldb_ + ncol] = (__bf16)v;
        }
    }
}

// ------------------------------------------------------------- LSTM pointwise
__global__ void lstm_pointwise_kernel(const float* __restrict__ gates,
                                      float* __restrict__ cF, float* __restrict__ hF,
                                      __bf16* __restrict__ hB,
                                      float* __restrict__ h_mirror,
                                      float* __restrict__ c_mirror) {
    int i = blockIdx.x * blockDim.x + threadIdx.x;      // B*H
    int b = i >> 10;
    int j = i & (H - 1);
    const float* g = gates + (size_t)b * 4 * H;
    float ig = sigmoidf_(g[j]);
    float fg = sigmoidf_(g[H + j]);
    float gg = tanhf(g[2 * H + j]);
    float og = sigmoidf_(g[3 * H + j]);
    float c = fg * cF[i] + ig * gg;
    float h = og * tanhf(c);
    cF[i] = c;
    hF[i] = h;
    hB[i] = (__bf16)h;
    h_mirror[i] = h;
    c_mirror[i] = c;
}

// ------------------------------------------------- attention scores (streaming)
// scores[b,s] = dot(energy[b,:], enc[b,s,:]).  One wave per s, 8 s per block.
template <typename T>
__global__ __launch_bounds__(256)
void attn_scores_kernel(const float* __restrict__ energy, const T* __restrict__ enc,
                        float* __restrict__ scores) {
    __shared__ float eLds[H];
    int b   = blockIdx.x / (S / 8);
    int sc  = blockIdx.x % (S / 8);
    int tid = threadIdx.x;
    for (int k = tid; k < H; k += 256) eLds[k] = energy[(size_t)b * H + k];
    __syncthreads();

    int wv   = tid >> 5;
    int lane = tid & 31;
    int s    = sc * 8 + wv;
    const T* row = enc + ((size_t)b * S + s) * H;
    // speculative prefetch 8 rows ahead (dropped silently if past buffer end)
    __builtin_prefetch(row + (size_t)8 * H + lane * 32, 0, 0);
    float sum = 0.0f;
    for (int h = lane; h < H; h += 32) sum += eLds[h] * (float)row[h];
#pragma unroll
    for (int off = 16; off > 0; off >>= 1) sum += __shfl_xor(sum, off, 32);
    if (lane == 0) scores[(size_t)b * S + s] = sum;
}

// ------------------------------------------------------------------- softmax
__global__ __launch_bounds__(256)
void softmax_kernel(const float* __restrict__ scores, float* __restrict__ weights,
                    float* __restrict__ w_out) {
    __shared__ float red[256];
    int b = blockIdx.x, tid = threadIdx.x;
    const float* row = scores + (size_t)b * S;

    float m = -3.4e38f;
    for (int s = tid; s < S; s += 256) m = fmaxf(m, row[s]);
    red[tid] = m; __syncthreads();
    for (int off = 128; off > 0; off >>= 1) {
        if (tid < off) red[tid] = fmaxf(red[tid], red[tid + off]);
        __syncthreads();
    }
    m = red[0]; __syncthreads();

    float sum = 0.0f;
    for (int s = tid; s < S; s += 256) {
        float e = expf(row[s] - m);
        weights[(size_t)b * S + s] = e;
        sum += e;
    }
    red[tid] = sum; __syncthreads();
    for (int off = 128; off > 0; off >>= 1) {
        if (tid < off) red[tid] += red[tid + off];
        __syncthreads();
    }
    float inv = 1.0f / red[0];
    for (int s = tid; s < S; s += 256) {
        float w = weights[(size_t)b * S + s] * inv;
        weights[(size_t)b * S + s] = w;
        w_out[(size_t)b * S + s]   = w;     // last step's values survive
    }
}

// ------------------------------------------------ attention context (streaming)
// ctx[b,h] = sum_s w[b,s] * enc[b,s,h].  256 h-columns per block.
// When available, enc tiles are double-buffered into LDS with
// GLOBAL_LOAD_ASYNC_TO_LDS_B128 (ASYNCcnt) to overlap streaming with compute.
template <typename T>
__global__ __launch_bounds__(256)
void attn_context_kernel(const float* __restrict__ weights, const T* __restrict__ enc,
                         __bf16* __restrict__ ctxB) {
    __shared__ float wLds[S];
    int b   = blockIdx.x >> 2;
    int h0  = (blockIdx.x & 3) * 256;
    int tid = threadIdx.x;
    wLds[tid]       = weights[(size_t)b * S + tid];
    wLds[tid + 256] = weights[(size_t)b * S + tid + 256];
    __syncthreads();

    const T* base = enc + (size_t)b * S * H + h0;       // this block's column window
    float acc = 0.0f;

#if HAS_ASYNC_LDS
    constexpr int TS = 8;                               // s-rows per tile
    constexpr int ELEMS_PER_CHUNK = 16 / sizeof(T);     // b128 chunks
    constexpr int CHUNKS   = (TS * 256) / ELEMS_PER_CHUNK;
    constexpr int PER_WAVE = (CHUNKS + 255) / 256;      // async ops per wave per tile
    __shared__ alignas(16) T tile[2][TS * 256];

    auto issue_tile = [&](int tblk, int buf) {
        for (int c = tid; c < CHUNKS; c += 256) {
            int e   = c * ELEMS_PER_CHUNK;
            int sl  = e >> 8;                           // s within tile
            int col = e & 255;                          // h within window
            const T* g = base + (size_t)(tblk * TS + sl) * H + col;
            __builtin_amdgcn_global_load_async_to_lds_b128(
                (gbl_i32x4*)(uintptr_t)g,
                (lds_i32x4*)(uintptr_t)&tile[buf][e], 0, 0);
        }
    };

    int buf = 0;
    issue_tile(0, 0);
    for (int tblk = 0; tblk < S / TS; ++tblk) {
        if (tblk + 1 < S / TS) {
            issue_tile(tblk + 1, buf ^ 1);
            __builtin_amdgcn_s_wait_asynccnt(PER_WAVE); // current tile landed
        } else {
            __builtin_amdgcn_s_wait_asynccnt(0);
        }
        __syncthreads();                                // all waves' copies visible
#pragma unroll
        for (int sl = 0; sl < TS; ++sl)
            acc += wLds[tblk * TS + sl] * (float)tile[buf][sl * 256 + tid];
        __syncthreads();                                // done reading before reuse
        buf ^= 1;
    }
#else
    for (int s = 0; s < S; ++s) {
        if (s + 8 < S) __builtin_prefetch(&base[(size_t)(s + 8) * H + tid], 0, 0);
        acc += wLds[s] * (float)base[(size_t)s * H + tid];
    }
#endif
    ctxB[(size_t)b * H + h0 + tid] = (__bf16)acc;
}

// -------------------------------------------------------------------- argmax
__global__ __launch_bounds__(256)
void argmax_kernel(const float* __restrict__ logits, int ld, int* __restrict__ inp) {
    __shared__ float bv[256];
    __shared__ int   bidx[256];
    int b = blockIdx.x, tid = threadIdx.x;
    const float* row = logits + (size_t)b * ld;
    float best = -3.4e38f; int bi = 0;
    for (int v = tid; v < V; v += 256) {
        float x = row[v];
        if (x > best) { best = x; bi = v; }
    }
    bv[tid] = best; bidx[tid] = bi; __syncthreads();
    for (int off = 128; off > 0; off >>= 1) {
        if (tid < off) {
            if (bv[tid + off] > bv[tid] ||
                (bv[tid + off] == bv[tid] && bidx[tid + off] < bidx[tid])) {
                bv[tid] = bv[tid + off]; bidx[tid] = bidx[tid + off];
            }
        }
        __syncthreads();
    }
    if (tid == 0) inp[b] = bidx[0];
}

// ============================================================== host launcher
extern "C" void kernel_launch(void* const* d_in, const int* in_sizes, int n_in,
                              void* d_out, int out_size, void* d_ws, size_t ws_size,
                              hipStream_t stream) {
    const float* enc      = (const float*)d_in[0];
    const float* h0       = (const float*)d_in[1];
    const float* c0       = (const float*)d_in[2];
    const float* emb      = (const float*)d_in[3];
    const float* W_ih     = (const float*)d_in[4];
    const float* W_hh     = (const float*)d_in[5];
    const float* b_ih     = (const float*)d_in[6];
    const float* b_hh     = (const float*)d_in[7];
    const float* attn_W   = (const float*)d_in[8];
    const float* attn_b   = (const float*)d_in[9];
    const float* concat_W = (const float*)d_in[10];
    const float* concat_b = (const float*)d_in[11];
    const float* lin_W    = (const float*)d_in[12];
    const float* lin_b    = (const float*)d_in[13];

    float* out   = (float*)d_out;
    float* out_h = out + (size_t)B * NSTEPS * V;
    float* out_c = out_h + (size_t)B * H;
    float* out_w = out_c + (size_t)B * H;

    // -------- workspace bump allocator (256B aligned)
    size_t off = 0;
    auto alloc = [&](size_t bytes) -> void* {
        size_t cur = (off + 255) & ~(size_t)255;
        off = cur + bytes;
        return (void*)((char*)d_ws + cur);
    };
    __bf16* W_ih_b   = (__bf16*)alloc((size_t)4 * H * H * 2);
    __bf16* W_hh_b   = (__bf16*)alloc((size_t)4 * H * H * 2);
    __bf16* attn_W_b = (__bf16*)alloc((size_t)H * H * 2);
    __bf16* cat_W_b  = (__bf16*)alloc((size_t)H * 2 * H * 2);
    __bf16* lin_W_b  = (__bf16*)alloc((size_t)V * H * 2);
    __bf16* x_b      = (__bf16*)alloc((size_t)B * H * 2);
    __bf16* h_b      = (__bf16*)alloc((size_t)B * H * 2);
    __bf16* ctx_b    = (__bf16*)alloc((size_t)B * H * 2);
    __bf16* co_b     = (__bf16*)alloc((size_t)B * H * 2);
    float*  gates    = (float*) alloc((size_t)B * 4 * H * 4);
    float*  hF       = (float*) alloc((size_t)B * H * 4);
    float*  cF       = (float*) alloc((size_t)B * H * 4);
    float*  energy   = (float*) alloc((size_t)B * H * 4);
    float*  scores   = (float*) alloc((size_t)B * S * 4);
    float*  weights  = (float*) alloc((size_t)B * S * 4);
    int*    inp      = (int*)   alloc((size_t)B * 4);
    __bf16* enc_b    = (__bf16*)alloc((size_t)B * S * H * 2);
    bool use_bf16_enc = (off <= ws_size);   // deterministic for a given ws_size

    auto cvt = [&](const float* src, __bf16* dst, int n) {
        f32_to_bf16_kernel<<<(n + 255) / 256, 256, 0, stream>>>(src, dst, n);
    };
    cvt(W_ih, W_ih_b, 4 * H * H);
    cvt(W_hh, W_hh_b, 4 * H * H);
    cvt(attn_W, attn_W_b, H * H);
    cvt(concat_W, cat_W_b, H * 2 * H);
    cvt(lin_W, lin_W_b, V * H);
    if (use_bf16_enc) cvt(enc, enc_b, B * S * H);

    init_state_kernel<<<(B * H) / 256, 256, 0, stream>>>(h0, c0, hF, cF, h_b);

    const int LD_OUT = NSTEPS * V;
    for (int t = 0; t < NSTEPS; ++t) {
        // 1. embedding lookup -> x (bf16)
        embed_kernel<<<(B * H) / 256, 256, 0, stream>>>(emb, inp, t == 0 ? 1 : 0, x_b);

        // 2. gates = x@W_ih^T + h@W_hh^T + b_ih + b_hh   [B,4H]
        wmma_gemm_bf16<<<(4 * H) / 32, 256, 0, stream>>>(
            x_b, H, W_ih_b, H, H,  h_b, H, W_hh_b, H, H,
            b_ih, b_hh, gates, 4 * H, nullptr, 0, 4 * H, 0);

        // 3. LSTM cell pointwise (also mirrors h,c into d_out)
        lstm_pointwise_kernel<<<(B * H) / 256, 256, 0, stream>>>(
            gates, cF, hF, h_b, out_h, out_c);

        // 4. energy = h@attn_W^T + attn_b   [B,H]
        wmma_gemm_bf16<<<H / 32, 256, 0, stream>>>(
            h_b, H, attn_W_b, H, H,  nullptr, 0, nullptr, 0, 0,
            attn_b, nullptr, energy, H, nullptr, 0, H, 0);

        // 5. scores[b,s] = energy[b]·enc[b,s]
        if (use_bf16_enc)
            attn_scores_kernel<__bf16><<<B * (S / 8), 256, 0, stream>>>(energy, enc_b, scores);
        else
            attn_scores_kernel<float><<<B * (S / 8), 256, 0, stream>>>(energy, enc, scores);

        // 6. softmax over S (writes ws + final-output mirror)
        softmax_kernel<<<B, 256, 0, stream>>>(scores, weights, out_w);

        // 7. context[b,:] = sum_s w[b,s]*enc[b,s,:]  -> bf16
        if (use_bf16_enc)
            attn_context_kernel<__bf16><<<B * 4, 256, 0, stream>>>(weights, enc_b, ctx_b);
        else
            attn_context_kernel<float><<<B * 4, 256, 0, stream>>>(weights, enc, ctx_b);

        // 8. co = tanh([h|ctx]@concat_W^T + concat_b)  [B,H] -> bf16
        wmma_gemm_bf16<<<H / 32, 256, 0, stream>>>(
            h_b, H, cat_W_b, 2 * H, H,  ctx_b, H, cat_W_b + H, 2 * H, H,
            concat_b, nullptr, nullptr, 0, co_b, H, H, 1);

        // 9. logits = co@lin_W^T + lin_b -> d_out[:, t, :]
        wmma_gemm_bf16<<<V / 32, 256, 0, stream>>>(
            co_b, H, lin_W_b, H, H,  nullptr, 0, nullptr, 0, 0,
            lin_b, nullptr, out + (size_t)t * V, LD_OUT, nullptr, 0, V, 0);

        // 10. greedy next token
        argmax_kernel<<<B, 256, 0, stream>>>(out + (size_t)t * V, LD_OUT, inp);
    }
}
